// Prototype_30820685316154
// MI455X (gfx1250) — compile-verified, matrix operand order checked by reading, not run on hardware
//
#include <hip/hip_runtime.h>

// ---------------------------------------------------------------------------
// Prototype-distance classifier, fused for MI455X (gfx1250, wave32, WMMA).
//   d2  = |x|^2 + |p|^2 - 2 x.p      (16384x1024 @ 1024x1024 bf16 WMMA GEMM)
//   sim = exp(-d2)                    (in registers)
//   logits = sim @ W^T + b            (second bf16 WMMA GEMM, C=16)
//   out = softmax(logits)
// One workgroup = 64 rows x all 1024 prototypes -> sim never hits HBM.
// ---------------------------------------------------------------------------

typedef __attribute__((ext_vector_type(16))) __bf16        v16bf;
typedef __attribute__((ext_vector_type(8)))  __bf16        v8bf;
typedef __attribute__((ext_vector_type(8)))  float         v8f;
typedef __attribute__((ext_vector_type(4)))  float         v4f;
typedef __attribute__((ext_vector_type(4)))  unsigned int  v4u;
typedef __attribute__((ext_vector_type(4)))  unsigned short v4us;

#define N_ROWS 16384   // B*T
#define H      1024
#define P_DIM  1024
#define C_DIM  16
#define M_BLK  64      // rows per workgroup
#define PCHUNK 256     // prototype columns per outer iteration
#define KSTEP  32      // bf16 WMMA K
#define AP     1032    // LDS pitch for A (elements), padded: 1032*2 B = 129*16
#define BP     40      // LDS pitch for B chunk (elements), padded: 80 B = 5*16
#define SIMP   72      // LDS pitch for sim staging tile (elements)

#define LDS_A_BYTES   (M_BLK * AP * 2)              // 132096
#define LDS_B_BYTES   (2 * PCHUNK * BP * 2)         // 40960 (sim stage overlays this)
#define LDS_L_BYTES   (M_BLK * C_DIM * 4)           // 4096
#define LDS_X_BYTES   (M_BLK * 4)                   // 256
#define LDS_TOTAL     (LDS_A_BYTES + LDS_B_BYTES + LDS_L_BYTES + LDS_X_BYTES)

__device__ __forceinline__ unsigned short f32_to_bf16(float f) {
  unsigned int u = __float_as_uint(f);
  u += 0x7FFFu + ((u >> 16) & 1u);   // round-to-nearest-even
  return (unsigned short)(u >> 16);
}

// Build a 16-element bf16 fragment from two contiguous 16-byte chunks
// (K-halves at +0 and +16 elements) -> two ds_load_b128 / global_load_b128.
__device__ __forceinline__ v16bf make_frag(const unsigned short* p) {
  v8bf lo = *(const v8bf*)(p);
  v8bf hi = *(const v8bf*)(p + 16);
  return __builtin_shufflevector(lo, hi, 0,1,2,3,4,5,6,7,8,9,10,11,12,13,14,15);
}

// ---------------------------------------------------------------------------
// Prep: one block per row of length 1024: f32 -> bf16 + row sum-of-squares.
// Used for prototypes [P,H] and for W [C,P] (P == H == 1024).
// ---------------------------------------------------------------------------
__global__ void __launch_bounds__(256)
row_prep_kernel(const float* __restrict__ src, unsigned short* __restrict__ dst,
                float* __restrict__ sq) {
  __shared__ float ssum;
  const int row = blockIdx.x;
  const int t   = threadIdx.x;
  if (t == 0) ssum = 0.0f;
  __syncthreads();
  v4f x = *(const v4f*)(src + row * H + t * 4);
  float local = x[0]*x[0] + x[1]*x[1] + x[2]*x[2] + x[3]*x[3];
  v4us o;
  o[0] = f32_to_bf16(x[0]); o[1] = f32_to_bf16(x[1]);
  o[2] = f32_to_bf16(x[2]); o[3] = f32_to_bf16(x[3]);
  *(v4us*)(dst + row * H + t * 4) = o;
  atomicAdd(&ssum, local);
  __syncthreads();
  if (t == 0) sq[row] = ssum;
}

// ---------------------------------------------------------------------------
// Main fused kernel. Grid: N_ROWS/M_BLK = 256 blocks x 256 threads (8 waves).
// ---------------------------------------------------------------------------
__global__ void __launch_bounds__(256)
proto_sim_kernel(const float* __restrict__ X,
                 const unsigned short* __restrict__ Pbf,
                 const unsigned short* __restrict__ Wbf,
                 const float* __restrict__ psq,
                 const float* __restrict__ bias,
                 float* __restrict__ out) {
  extern __shared__ char smem[];
  unsigned short* As      = (unsigned short*)(smem);
  unsigned short* Bs      = (unsigned short*)(smem + LDS_A_BYTES);
  unsigned short* simS    = Bs;  // overlays B buffers (dead during 2nd GEMM)
  float*          logitsS = (float*)(smem + LDS_A_BYTES + LDS_B_BYTES);
  float*          xsqS    = (float*)(smem + LDS_A_BYTES + LDS_B_BYTES + LDS_L_BYTES);

  const int t    = threadIdx.x;
  const int lane = t & 31;
  const int wid  = t >> 5;
  const int l15  = lane & 15;
  const int lh   = lane >> 4;
  const int mw   = wid & 1;    // 2 waves along M
  const int nw   = wid >> 1;   // 4 waves along N (prototype cols)
  const int rowBase = blockIdx.x * M_BLK;

  // ---- init shared accumulators -------------------------------------------
  for (int i = t; i < M_BLK * C_DIM; i += 256) logitsS[i] = 0.0f;
  if (t < M_BLK) xsqS[t] = 0.0f;
  __syncthreads();

  // ---- stage X block into LDS as bf16 + |x|^2 -----------------------------
  {
    const int arow = t >> 2;                 // 0..63
    const int aq   = t & 3;                  // quarter of the row
    const float* srcRow = X + (size_t)(rowBase + arow) * H + aq * 256;
    unsigned short* dstRow = As + arow * AP + aq * 256;
    float acc = 0.0f;
#pragma unroll 4
    for (int j = 0; j < 64; ++j) {
      v4f x = *(const v4f*)(srcRow + j * 4);
      acc += x[0]*x[0] + x[1]*x[1] + x[2]*x[2] + x[3]*x[3];
      v4us o;
      o[0] = f32_to_bf16(x[0]); o[1] = f32_to_bf16(x[1]);
      o[2] = f32_to_bf16(x[2]); o[3] = f32_to_bf16(x[3]);
      *(v4us*)(dstRow + j * 4) = o;
    }
    atomicAdd(&xsqS[arow], acc);
  }
  __syncthreads();

  // Per-lane |x|^2 for this wave's accumulator rows (D-layout: M = v + 8*lh).
  float xsqr[16];
#pragma unroll
  for (int rt = 0; rt < 2; ++rt)
#pragma unroll
    for (int v = 0; v < 8; ++v)
      xsqr[rt * 8 + v] = xsqS[mw * 32 + rt * 16 + v + 8 * lh];

  const v8f vzero = {0.f,0.f,0.f,0.f,0.f,0.f,0.f,0.f};
  const int r0 = t >> 2;     // staging: row-in-chunk base
  const int kq = t & 3;      // staging: 16B segment within 32-elem K slab

  // ---- loop over prototype column chunks ----------------------------------
  for (int pc = 0; pc < P_DIM / PCHUNK; ++pc) {
    const int pcBase = pc * PCHUNK;

    float psqv[4];
#pragma unroll
    for (int ct = 0; ct < 4; ++ct)
      psqv[ct] = psq[pcBase + nw * 64 + ct * 16 + l15];

    v8f acc[2][4];
#pragma unroll
    for (int rt = 0; rt < 2; ++rt)
#pragma unroll
      for (int ct = 0; ct < 4; ++ct) acc[rt][ct] = vzero;

    __syncthreads();   // sim-stage reads from previous chunk complete

    // prime B double-buffer with k=0 slab (256 rows x 32 bf16)
#pragma unroll
    for (int j = 0; j < 4; ++j) {
      v4u d = *(const v4u*)(Pbf + (size_t)(pcBase + 64 * j + r0) * H + kq * 8);
      *(v4u*)(Bs + (64 * j + r0) * BP + kq * 8) = d;
    }
    __syncthreads();

    // ---- K loop: double-buffered LDS staging + 8 WMMA per step ------------
    for (int ks = 0; ks < H / KSTEP; ++ks) {
      const int cur = ks & 1;
      v4u pre[4];
      const bool hasNext = (ks + 1) < (H / KSTEP);
      if (hasNext) {
        const int kn = (ks + 1) * KSTEP;
#pragma unroll
        for (int j = 0; j < 4; ++j)
          pre[j] = *(const v4u*)(Pbf + (size_t)(pcBase + 64 * j + r0) * H + kn + kq * 8);
      }

      const unsigned short* bsc = Bs + cur * (PCHUNK * BP);
      v16bf afrag[2];
#pragma unroll
      for (int rt = 0; rt < 2; ++rt)
        afrag[rt] = make_frag(As + (mw * 32 + rt * 16 + l15) * AP + ks * KSTEP + lh * 8);

#pragma unroll
      for (int ct = 0; ct < 4; ++ct) {
        v16bf bfrag = make_frag(bsc + (nw * 64 + ct * 16 + l15) * BP + lh * 8);
#pragma unroll
        for (int rt = 0; rt < 2; ++rt)
          acc[rt][ct] = __builtin_amdgcn_wmma_f32_16x16x32_bf16(
              false, afrag[rt], false, bfrag, (short)0, acc[rt][ct], false, false);
      }

      if (hasNext) {
        unsigned short* bsn = Bs + (cur ^ 1) * (PCHUNK * BP);
#pragma unroll
        for (int j = 0; j < 4; ++j)
          *(v4u*)(bsn + (64 * j + r0) * BP + kq * 8) = pre[j];
      }
      __syncthreads();
    }

    // ---- sim = exp(2*dot - |x|^2 - |p|^2), restage as bf16 (A-layout src) --
    unsigned short* sw = simS + wid * (32 * SIMP);
#pragma unroll
    for (int rt = 0; rt < 2; ++rt)
#pragma unroll
      for (int ct = 0; ct < 4; ++ct)
#pragma unroll
        for (int v = 0; v < 8; ++v) {
          float simv = __expf(2.0f * acc[rt][ct][v] - xsqr[rt * 8 + v] - psqv[ct]);
          sw[(rt * 16 + v + 8 * lh) * SIMP + ct * 16 + l15] = f32_to_bf16(simv);
        }
    // same-wave LDS ops are in-order: no barrier needed before re-reading sw

    // ---- second GEMM: logits partial = sim(16x64) @ W^T(64x16) ------------
#pragma unroll
    for (int rt = 0; rt < 2; ++rt) {
      v8f plog = vzero;
#pragma unroll
      for (int kc = 0; kc < 2; ++kc) {
        v16bf af = make_frag(sw + (rt * 16 + l15) * SIMP + kc * 32 + lh * 8);
        v16bf wf = make_frag(Wbf + (size_t)l15 * P_DIM + pcBase + nw * 64 + kc * 32 + lh * 8);
        plog = __builtin_amdgcn_wmma_f32_16x16x32_bf16(
            false, af, false, wf, (short)0, plog, false, false);
      }
#pragma unroll
      for (int v = 0; v < 8; ++v)
        atomicAdd(&logitsS[(mw * 32 + rt * 16 + v + 8 * lh) * C_DIM + l15], plog[v]);
    }
  }

  __syncthreads();

  // ---- bias + softmax over C=16, one thread per row -----------------------
  if (t < M_BLK) {
    float lg[C_DIM];
    float m = -3.0e38f;
#pragma unroll
    for (int c = 0; c < C_DIM; ++c) {
      lg[c] = logitsS[t * C_DIM + c] + bias[c];
      m = fmaxf(m, lg[c]);
    }
    float s = 0.0f;
#pragma unroll
    for (int c = 0; c < C_DIM; ++c) { lg[c] = __expf(lg[c] - m); s += lg[c]; }
    const float inv = 1.0f / s;
    float* op = out + (size_t)(rowBase + t) * C_DIM;
#pragma unroll
    for (int c = 0; c < C_DIM; ++c) op[c] = lg[c] * inv;
  }
}

// ---------------------------------------------------------------------------
extern "C" void kernel_launch(void* const* d_in, const int* in_sizes, int n_in,
                              void* d_out, int out_size, void* d_ws, size_t ws_size,
                              hipStream_t stream) {
  const float* X     = (const float*)d_in[0];   // [B,T,H] f32
  const float* proto = (const float*)d_in[1];   // [P,H]   f32
  const float* W     = (const float*)d_in[2];   // [C,P]   f32
  const float* b     = (const float*)d_in[3];   // [C]     f32
  float* out = (float*)d_out;                   // [B,T,C] f32

  // Workspace carve (~2.14 MB): bf16 prototypes, bf16 W, |p|^2, dummy sq.
  char* ws = (char*)d_ws;
  unsigned short* Pbf = (unsigned short*)(ws);
  unsigned short* Wbf = (unsigned short*)(ws + (size_t)P_DIM * H * 2);
  float* psq          = (float*)(ws + (size_t)P_DIM * H * 2 + (size_t)C_DIM * P_DIM * 2);
  float* wsq          = psq + P_DIM;   // unused sums for W rows

  row_prep_kernel<<<P_DIM, 256, 0, stream>>>(proto, Pbf, psq);
  row_prep_kernel<<<C_DIM, 256, 0, stream>>>(W, Wbf, wsq);  // W rows len P==1024

  proto_sim_kernel<<<N_ROWS / M_BLK, 256, LDS_TOTAL, stream>>>(
      X, Pbf, Wbf, psq, b, out);
}
